// Attention_25512105738495
// MI455X (gfx1250) — matrix operand order repaired and checked by reading, not compiled, not run on hardware
//
#include <hip/hip_runtime.h>

#define HIDDEN    4096
#define NUM_HEADS 32
#define HEAD_DIM  128
#define MAX_SEQ   4096
#define BATCH     8
#define SEQ       16
#define MROWS     (BATCH * SEQ)   // 128

typedef __attribute__((ext_vector_type(16))) _Float16 v16h;
typedef __attribute__((ext_vector_type(8)))  _Float16 v8h;
typedef __attribute__((ext_vector_type(8)))  float    v8f;
typedef __attribute__((ext_vector_type(4)))  int      v4i;

#if defined(__has_builtin)
#if __has_builtin(__builtin_amdgcn_global_load_async_to_lds_b128)
#define HAVE_ASYNC_LDS 1
#endif
#endif
#ifndef HAVE_ASYNC_LDS
#define HAVE_ASYNC_LDS 0
#endif

#if HAVE_ASYNC_LDS
typedef __attribute__((address_space(1))) v4i* gv4i_p;   // global int4*
typedef __attribute__((address_space(3))) v4i* lv4i_p;   // LDS int4*
#endif

// ---------------------------------------------------------------------------
// WMMA helper:  D(16x16 f32) = A(16x32 f16) * B(32x16 f16) + C
static __device__ __forceinline__ v8f wmma_f16(v16h a, v16h b, v8f c) {
  return __builtin_amdgcn_wmma_f32_16x16x32_f16(
      /*neg_a=*/false, a, /*neg_b=*/false, b,
      /*c_mod=*/(short)0, c, /*reuse_a=*/false, /*reuse_b=*/false);
}

// B-operand pack: per-lane 16 consecutive f32 -> f16.
// (B layout: lane r holds column n=r; elem e holds K = hi*16 + e)
static __device__ __forceinline__ v16h packB(const float* p) {
  v16h r;
#pragma unroll
  for (int i = 0; i < 16; ++i) r[i] = (_Float16)p[i];
  return r;
}

// A-operand pack: two runs of 8 consecutive f32.
// (A layout: lane r holds row m=r; elems 0..7 -> K = hi*16+i, elems 8..15 -> K = hi*16+8+i)
static __device__ __forceinline__ v16h packA(const float* p0, const float* p1) {
  v16h r;
#pragma unroll
  for (int i = 0; i < 8; ++i) { r[i] = (_Float16)p0[i]; r[8 + i] = (_Float16)p1[i]; }
  return r;
}

// ---------------------------------------------------------------------------
// GEMM:  C[m][n] = sum_k A[m][k] * W[n][k]      (i.e. C = A @ W^T)
// A: MxK row-major f32, W: NxK row-major f32, C: MxN f32.
// Block = 64 threads (2 waves); each wave computes a 16x64 tile (4 n-tiles,
// A operand reused across them). Grid: (N/128, M/16).
__global__ void __launch_bounds__(64)
gemm_xwT(const float* __restrict__ A, const float* __restrict__ W,
         float* __restrict__ C, int M, int N, int K) {
  const int lane = threadIdx.x & 31;
  const int wid  = threadIdx.x >> 5;
  const int r  = lane & 15;
  const int hi = lane >> 4;
  const int m0 = blockIdx.y * 16;
  const int n0 = blockIdx.x * 128 + wid * 64;

  v8f acc[4] = {};
  const float* arow = A + (long)(m0 + r) * K;

  for (int k0 = 0; k0 < K; k0 += 32) {
    v16h av = packA(arow + k0 + hi * 16, arow + k0 + hi * 16 + 8);
#pragma unroll
    for (int nt = 0; nt < 4; ++nt) {
      const float* wrow = W + (long)(n0 + nt * 16 + r) * K + k0;
      v16h bv = packB(wrow + hi * 16);
      acc[nt] = wmma_f16(av, bv, acc[nt]);
    }
  }
  // C layout: lane r = column, vgpr v = row (v + 8*hi)
#pragma unroll
  for (int nt = 0; nt < 4; ++nt)
#pragma unroll
    for (int v = 0; v < 8; ++v)
      C[(long)(m0 + v + 8 * hi) * N + n0 + nt * 16 + r] = acc[nt][v];
}

// ---------------------------------------------------------------------------
// Online-softmax update + P@V accumulate for one 32-token chunk.
// s0/s1: two 16x16 score tiles in WMMA C layout (row = v+8*hi, col = r).
// V rows j (0..31) at v0base + j*vstride + (n-col); rows >= nvalid read as 0.
__device__ __forceinline__ void softmax_pv(
    v8f s0, v8f s1, const float* __restrict__ v0base, long vstride, int nvalid,
    int r, int hi, float mrow[8], float lrow[8], v8f acc[8],
    _Float16 (*P)[32]) {
  float p0[8], p1[8];
#pragma unroll
  for (int v = 0; v < 8; ++v) {
    float x = fmaxf(s0[v], s1[v]);
    x = fmaxf(x, __shfl_xor(x, 1));
    x = fmaxf(x, __shfl_xor(x, 2));
    x = fmaxf(x, __shfl_xor(x, 4));
    x = fmaxf(x, __shfl_xor(x, 8));
    float mn    = fmaxf(mrow[v], x);
    float alpha = __expf(mrow[v] - mn);
    mrow[v] = mn;
    p0[v] = __expf(s0[v] - mn);
    p1[v] = __expf(s1[v] - mn);
    float rs = p0[v] + p1[v];
    rs += __shfl_xor(rs, 1);
    rs += __shfl_xor(rs, 2);
    rs += __shfl_xor(rs, 4);
    rs += __shfl_xor(rs, 8);
    lrow[v] = lrow[v] * alpha + rs;
#pragma unroll
    for (int nt = 0; nt < 8; ++nt) acc[nt][v] *= alpha;
  }
  // relayout P (C layout) -> A layout through LDS (per-wave 16x32 f16 tile)
#pragma unroll
  for (int v = 0; v < 8; ++v) {
    P[v + 8 * hi][r]      = (_Float16)p0[v];
    P[v + 8 * hi][16 + r] = (_Float16)p1[v];
  }
  asm volatile("s_wait_dscnt 0" ::: "memory");  // same-wave cross-lane LDS RAW
  v16h pa;
  {
    v8h a0 = *(const v8h*)&P[r][hi * 16];
    v8h a1 = *(const v8h*)&P[r][hi * 16 + 8];
#pragma unroll
    for (int i = 0; i < 8; ++i) { pa[i] = a0[i]; pa[8 + i] = a1[i]; }
  }
  // O(16x128) += P(16x32) @ V(32x128), 8 n-tiles
#pragma unroll
  for (int nt = 0; nt < 8; ++nt) {
    v16h bv;
#pragma unroll
    for (int e = 0; e < 16; ++e) {
      int j = hi * 16 + e;
      bv[e] = (j < nvalid) ? (_Float16)v0base[(long)j * vstride + nt * 16 + r]
                           : (_Float16)0.0f;
    }
    acc[nt] = wmma_f16(pa, bv, acc[nt]);
  }
}

// ---------------------------------------------------------------------------
// Attention: one block per (b, head); 2 waves split t-chunks flash-style.
// Each wave stages its V chunk (32x128 f32, 16KB) into LDS with CDNA5 async
// global->LDS copies (ASYNCcnt), overlapped with the Q@K^T WMMAs.
__global__ void __launch_bounds__(64)
attn_kernel(const float* __restrict__ Q,      // 128 x 4096 (workspace)
            const float* __restrict__ Knew,   // 128 x 4096 (workspace)
            const float* __restrict__ Vnew,   // 128 x 4096 (workspace)
            const float* __restrict__ Kc,     // B,H,MAX_SEQ,HEAD_DIM
            const float* __restrict__ Vc,
            const int*   __restrict__ posp,
            float* __restrict__ Oout) {       // 128 x 4096 (workspace)
  const int pos  = *posp;                     // assumed multiple of 32 (2048)
  const int bh   = blockIdx.x;
  const int b    = bh >> 5, h = bh & 31;
  const int lane = threadIdx.x & 31;
  const int wid  = threadIdx.x >> 5;          // 0..1
  const int r    = lane & 15;
  const int hi   = lane >> 4;

  __shared__ __align__(16) float    vstage[2][32][HEAD_DIM];   // 32 KB
  __shared__ __align__(16) _Float16 lds_p[2][16][32];          //  2 KB
  __shared__ float sm_m[2][16], sm_l[2][16];
  __shared__ float sm_o[2][16][HEAD_DIM];                      // 16 KB

  const float scale = 0.08838834764831845f;   // 1/sqrt(128), folded into Q
  const float* qrow = Q + (long)(b * SEQ + r) * HIDDEN + h * HEAD_DIM;
  v16h aq[4];
#pragma unroll
  for (int dc = 0; dc < 4; ++dc) {
    v16h t;
#pragma unroll
    for (int i = 0; i < 8; ++i) {
      t[i]     = (_Float16)(qrow[dc * 32 + hi * 16 + i] * scale);
      t[8 + i] = (_Float16)(qrow[dc * 32 + hi * 16 + 8 + i] * scale);
    }
    aq[dc] = t;
  }

  const float* kbase  = Kc + (long)bh * MAX_SEQ * HEAD_DIM;
  const float* vbase  = Vc + (long)bh * MAX_SEQ * HEAD_DIM;
  const float* KnewBH = Knew + (long)(b * SEQ) * HIDDEN + h * HEAD_DIM;
  const float* VnewBH = Vnew + (long)(b * SEQ) * HIDDEN + h * HEAD_DIM;

  float mrow[8], lrow[8];
  v8f acc[8] = {};
#pragma unroll
  for (int v = 0; v < 8; ++v) { mrow[v] = -1e30f; lrow[v] = 0.0f; }

  // ---- full 32-token chunks over the existing cache (t < pos, no mask) ----
  const int nfull = pos >> 5;
  for (int c = wid; c < nfull; c += 2) {
    const int t0 = c << 5;
#if HAVE_ASYNC_LDS
    // Previous iteration's ds reads of vstage must complete before the async
    // engine overwrites it (async LDS writes are unordered vs this wave's DS).
    asm volatile("s_wait_dscnt 0" ::: "memory");
    {
      const float* gsrc = vbase + (long)t0 * HEAD_DIM;   // 32x128 f32 chunk
      float* lb = &vstage[wid][0][0];
#pragma unroll
      for (int kk = 0; kk < 32; ++kk) {                  // 32 x (32 lanes x 16B)
        const int off = (kk * 32 + lane) * 4;
        __builtin_amdgcn_global_load_async_to_lds_b128(
            (gv4i_p)(gsrc + off), (lv4i_p)(lb + off), 0, 0);
      }
    }
#endif
    if (c + 2 < nfull)
      __builtin_prefetch(kbase + (long)(t0 + 64 + r) * HEAD_DIM, 0, 0);

    v8f s0 = {}, s1 = {};
    const float* k0r = kbase + (long)(t0 + r) * HEAD_DIM;  // B col n=r -> K row t0+r
    const float* k1r = k0r + 16 * HEAD_DIM;
#pragma unroll
    for (int dc = 0; dc < 4; ++dc) {
      s0 = wmma_f16(aq[dc], packB(k0r + dc * 32 + hi * 16), s0);
      s1 = wmma_f16(aq[dc], packB(k1r + dc * 32 + hi * 16), s1);
    }

#if HAVE_ASYNC_LDS
    asm volatile("s_wait_asynccnt 0" ::: "memory");        // V chunk landed
    const float* vptr = &vstage[wid][0][0];
#else
    const float* vptr = vbase + (long)t0 * HEAD_DIM;
#endif
    softmax_pv(s0, s1, vptr, HEAD_DIM, 32, r, hi, mrow, lrow, acc, lds_p[wid]);
  }

  // ---- tail: 16 freshly projected tokens (t in [pos, pos+16)), causal ----
  if (wid == 1) {
    v8f s0 = {}, s1;
    const float* knr = KnewBH + (long)r * HIDDEN;          // new-K row j=r
#pragma unroll
    for (int dc = 0; dc < 4; ++dc)
      s0 = wmma_f16(aq[dc], packB(knr + dc * 32 + hi * 16), s0);
#pragma unroll
    for (int v = 0; v < 8; ++v) {
      if (r > v + 8 * hi) s0[v] = -1e30f;  // col t=pos+r vs row m=v+8*hi
      s1[v] = -1e30f;                      // tile 1 entirely out of range
    }
#if HAVE_ASYNC_LDS
    asm volatile("s_wait_dscnt 0" ::: "memory");
#endif
    softmax_pv(s0, s1, VnewBH, HIDDEN, 16, r, hi, mrow, lrow, acc, lds_p[wid]);
  }

  // ---- merge the 2 waves' partial (m, l, O) ----
  if (r == 0) {
#pragma unroll
    for (int v = 0; v < 8; ++v) {
      sm_m[wid][v + 8 * hi] = mrow[v];
      sm_l[wid][v + 8 * hi] = lrow[v];
    }
  }
#pragma unroll
  for (int nt = 0; nt < 8; ++nt)
#pragma unroll
    for (int v = 0; v < 8; ++v)
      sm_o[wid][v + 8 * hi][nt * 16 + r] = acc[nt][v];
  __syncthreads();

  for (int idx = threadIdx.x; idx < SEQ * HEAD_DIM; idx += 64) {
    const int m = idx >> 7, n = idx & (HEAD_DIM - 1);
    float M = fmaxf(sm_m[0][m], sm_m[1][m]);
    float den = 0.0f, num = 0.0f;
#pragma unroll
    for (int w = 0; w < 2; ++w) {
      const float e = __expf(sm_m[w][m] - M);
      den += sm_l[w][m] * e;
      num += sm_o[w][m][n] * e;
    }
    Oout[(long)(b * SEQ + m) * HIDDEN + h * HEAD_DIM + n] = num / den;
  }
}

// ---------------------------------------------------------------------------
extern "C" void kernel_launch(void* const* d_in, const int* in_sizes, int n_in,
                              void* d_out, int out_size, void* d_ws, size_t ws_size,
                              hipStream_t stream) {
  (void)in_sizes; (void)n_in; (void)out_size; (void)ws_size;
  const float* h  = (const float*)d_in[0];
  const float* Wq = (const float*)d_in[1];
  const float* Wk = (const float*)d_in[2];
  const float* Wv = (const float*)d_in[3];
  const float* Wo = (const float*)d_in[4];
  const float* Kc = (const float*)d_in[5];
  const float* Vc = (const float*)d_in[6];
  const int*  pos = (const int*)d_in[7];
  float* out = (float*)d_out;

  float* wsQ = (float*)d_ws;                 // 128 x 4096
  float* wsK = wsQ + (size_t)MROWS * HIDDEN; // 128 x 4096
  float* wsV = wsK + (size_t)MROWS * HIDDEN; // 128 x 4096
  float* wsO = wsV + (size_t)MROWS * HIDDEN; // 128 x 4096

  dim3 gg(HIDDEN / 128, MROWS / 16);         // (32, 8)
  gemm_xwT<<<gg, 64, 0, stream>>>(h, Wq, wsQ, MROWS, HIDDEN, HIDDEN);
  gemm_xwT<<<gg, 64, 0, stream>>>(h, Wk, wsK, MROWS, HIDDEN, HIDDEN);
  gemm_xwT<<<gg, 64, 0, stream>>>(h, Wv, wsV, MROWS, HIDDEN, HIDDEN);

  attn_kernel<<<BATCH * NUM_HEADS, 64, 0, stream>>>(wsQ, wsK, wsV, Kc, Vc, pos, wsO);

  gemm_xwT<<<gg, 64, 0, stream>>>(wsO, Wo, out, MROWS, HIDDEN, HIDDEN);
}
